// VectorQuantizer_36086315221018
// MI455X (gfx1250) — matrix operand order repaired and checked by reading, not compiled, not run on hardware
//
#include <hip/hip_runtime.h>
#include <hip/hip_bf16.h>

// ---------------------------------------------------------------------------
// Problem constants (from reference): z [B=32, D=256, T=2048], emb [K=1024, D]
// ---------------------------------------------------------------------------
#define B_   32
#define D_   256
#define T_   2048
#define K_   1024
#define N_   (B_ * T_)            // 65536 rows
#define OUT_MAIN (B_ * D_ * T_)   // 16777216 floats

// main-kernel tiling: 128 rows / block, 4 waves, 32 rows per wave (2 A-tiles)
#define TILE_ROWS 128
#define WAVES     4
#define THREADS   (WAVES * 32)
#define ROWS_PER_WAVE 32
#define LDS_STR   264             // padded row stride in bf16 elems (528B = 16*33)
#define BTILE_SH  4096            // 16 codes * 256 d (ushorts) = 8KB per buffer

typedef __attribute__((ext_vector_type(16))) __bf16 v16bf;
typedef __attribute__((ext_vector_type(8)))  float  v8f;

union Frag16 {                    // 16 bf16 = one WMMA A/B fragment per lane
    v16bf v;
    uint4 q[2];
};

static __device__ __forceinline__ unsigned short f32_to_bf16_rne(float f) {
    unsigned int u = __float_as_uint(f);
    unsigned int r = (u + 0x7FFFu + ((u >> 16) & 1u)) >> 16;
    return (unsigned short)r;
}

// async copy of one 16-code bf16 tile (8KB) global -> LDS, 4 x b128 per thread
static __device__ __forceinline__ void stage_btile(const unsigned short* __restrict__ src,
                                                   unsigned ldsBase, int tid) {
    #pragma unroll
    for (int i = 0; i < 4; ++i) {
        int j = i * THREADS + tid;                 // 0..511 x 16B = 8KB
        unsigned lds = ldsBase + (unsigned)(j * 16);
        unsigned long long g = (unsigned long long)((const char*)src + j * 16);
        asm volatile("global_load_async_to_lds_b128 %0, %1, off"
                     :: "v"(lds), "v"(g) : "memory");
    }
}

static __device__ __forceinline__ void wait_async0() {
    asm volatile("s_wait_asynccnt 0" ::: "memory");
}

// ---------------------------------------------------------------------------
// 0) zero histogram + loss accumulator
// ---------------------------------------------------------------------------
__global__ void vq_init(float* counts, float* lossAcc) {
    int t = threadIdx.x;
    if (t < K_) counts[t] = 0.0f;
    if (t == 0) *lossAcc = 0.0f;
}

// ---------------------------------------------------------------------------
// 1) pack codebook to bf16 + per-code squared norm (f32)
// ---------------------------------------------------------------------------
__global__ void vq_prep_emb(const float* __restrict__ emb,
                            unsigned short* __restrict__ emb_bf,
                            float* __restrict__ e_norm) {
    int k = blockIdx.x;
    int d = threadIdx.x;
    float v = emb[k * D_ + d];
    emb_bf[k * D_ + d] = f32_to_bf16_rne(v);
    __shared__ float s[D_];
    s[d] = v * v;
    __syncthreads();
    for (int off = D_ / 2; off > 0; off >>= 1) {
        if (d < off) s[d] += s[d + off];
        __syncthreads();
    }
    if (d == 0) e_norm[k] = s[0];
}

// ---------------------------------------------------------------------------
// argmin emission for one 16-row C tile
// ---------------------------------------------------------------------------
static __device__ __forceinline__ void emit_rows(float* bestV, int* bestI,
                                                 float znRow, int rowBase, int lane,
                                                 int* __restrict__ idxOut,
                                                 float* __restrict__ counts,
                                                 float* sLoss) {
    for (int r = 0; r < 8; ++r) {
        float v0 = bestV[r];
        int   i0 = bestI[r];
        #pragma unroll
        for (int msk = 8; msk >= 1; msk >>= 1) {
            float ov = __shfl_xor(v0, msk);
            int   oi = __shfl_xor(i0, msk);
            if (ov < v0 || (ov == v0 && oi < i0)) { v0 = ov; i0 = oi; }
        }
        float znA = __shfl(znRow, r);        // norm of local row r
        float znB = __shfl(znRow, r + 8);    // norm of local row r+8
        if (lane == 0) {                     // winner for row r (lanes 0-15)
            idxOut[rowBase + r] = i0;
            atomicAdd(&counts[i0], 1.0f);
            atomicAdd(sLoss, v0 + znA);
        } else if (lane == 16) {             // winner for row r+8 (lanes 16-31)
            idxOut[rowBase + r + 8] = i0;
            atomicAdd(&counts[i0], 1.0f);
            atomicAdd(sLoss, v0 + znB);
        }
    }
}

// ---------------------------------------------------------------------------
// 2) main: WMMA distance matmul + argmin + histogram + loss
//    block = 128 rows (same b, consecutive t); wave owns 32 rows (2 A tiles);
//    B code-tiles double-buffered in LDS via async global->LDS copies
// ---------------------------------------------------------------------------
__global__ void __launch_bounds__(THREADS)
vq_main(const float* __restrict__ z,
        const unsigned short* __restrict__ emb_bf,
        const float* __restrict__ e_norm,
        int* __restrict__ idxOut,
        float* __restrict__ counts,
        float* __restrict__ lossAcc) {
    __shared__ __align__(16) unsigned short tile[TILE_ROWS * LDS_STR]; // zf bf16, transposed
    __shared__ __align__(16) unsigned short bBuf[2][BTILE_SH];         // 2 x 8KB B tiles
    __shared__ float ldsEN[K_];
    __shared__ float sLoss;

    const int tid  = threadIdx.x;
    const int lane = tid & 31;
    const int wv   = tid >> 5;
    const int half = lane >> 4;     // 0: lanes 0-15, 1: lanes 16-31
    const int m    = lane & 15;

    if (tid == 0) sLoss = 0.0f;

    // preload code norms into LDS
    for (int j = tid; j < K_; j += THREADS) ldsEN[j] = e_norm[j];

    // ---- stage z tile: read [d][t] coalesced, write bf16 [t][d] to LDS ----
    const int n0 = blockIdx.x * TILE_ROWS;      // global row base
    const int b  = n0 >> 11;                    // n0 / T_
    const int t0 = n0 & (T_ - 1);
    const long zBase = (long)b * D_ * T_ + t0;
    for (int it = 0; it < (TILE_ROWS * (D_ / 2)) / THREADS; ++it) {
        int linear = it * THREADS + tid;        // 0 .. 16383
        int t  = linear & (TILE_ROWS - 1);
        int dp = linear >> 7;                   // d pair index 0..127
        int d  = dp * 2;
        float v0 = z[zBase + (long)d * T_ + t];
        float v1 = z[zBase + (long)(d + 1) * T_ + t];
        unsigned int pack = (unsigned int)f32_to_bf16_rne(v0) |
                            ((unsigned int)f32_to_bf16_rne(v1) << 16);
        *(unsigned int*)&tile[t * LDS_STR + d] = pack;
    }

    // kick off async copy of first B tile while tile-barrier settles
    const unsigned bBase0 = (unsigned)(size_t)&bBuf[0][0];
    const unsigned bBase1 = (unsigned)(size_t)&bBuf[1][0];
    stage_btile(emb_bf, bBase0, tid);

    __syncthreads();

    // ---- build A fragments (ISA 16-bit A 16x32 layout) + row norms ----
    // a0: rows wv*32 + m ; a1: rows wv*32 + 16 + m
    // v16bf elems 0..7  -> K = half*8 + 0..7      (VGPRs 0-3)
    // v16bf elems 8..15 -> K = 16 + half*8 + 0..7 (VGPRs 4-7)
    Frag16 a0[8], a1[8];
    float zn0 = 0.0f, zn1 = 0.0f;
    {
        const int r0 = (wv * ROWS_PER_WAVE + m) * LDS_STR;
        const int r1 = (wv * ROWS_PER_WAVE + 16 + m) * LDS_STR;
        for (int c = 0; c < 8; ++c) {
            a0[c].q[0] = *(const uint4*)&tile[r0 + c * 32 + half * 8];
            a0[c].q[1] = *(const uint4*)&tile[r0 + c * 32 + 16 + half * 8];
            a1[c].q[0] = *(const uint4*)&tile[r1 + c * 32 + half * 8];
            a1[c].q[1] = *(const uint4*)&tile[r1 + c * 32 + 16 + half * 8];
            #pragma unroll
            for (int qi = 0; qi < 2; ++qi)
                #pragma unroll
                for (int e = 0; e < 4; ++e) {
                    unsigned int u0 = (&a0[c].q[qi].x)[e];
                    unsigned int u1 = (&a1[c].q[qi].x)[e];
                    float l0 = __uint_as_float(u0 << 16);
                    float h0 = __uint_as_float(u0 & 0xFFFF0000u);
                    float l1 = __uint_as_float(u1 << 16);
                    float h1 = __uint_as_float(u1 & 0xFFFF0000u);
                    zn0 += l0 * l0 + h0 * h0;
                    zn1 += l1 * l1 + h1 * h1;
                }
        }
    }
    float znRow0 = zn0 + __shfl_xor(zn0, 16);
    float znRow1 = zn1 + __shfl_xor(zn1, 16);

    // ---- loop over 64 code tiles, double-buffered B in LDS ----
    float bestV0[8], bestV1[8];
    int   bestI0[8], bestI1[8];
    #pragma unroll
    for (int r = 0; r < 8; ++r) {
        bestV0[r] = 3.0e38f; bestI0[r] = 0;
        bestV1[r] = 3.0e38f; bestI1[r] = 0;
    }

    int buf = 0;
    for (int ct = 0; ct < K_ / 16; ++ct) {
        wait_async0();                 // this wave's async copies done
        __syncthreads();               // everyone's copies + prev compute done
        if (ct + 1 < K_ / 16)
            stage_btile(emb_bf + (ct + 1) * 16 * D_,
                        buf ? bBase0 : bBase1, tid);

        const int code = ct * 16 + m;
        v8f acc0 = {0.f, 0.f, 0.f, 0.f, 0.f, 0.f, 0.f, 0.f};
        v8f acc1 = {0.f, 0.f, 0.f, 0.f, 0.f, 0.f, 0.f, 0.f};
        #pragma unroll
        for (int c = 0; c < 8; ++c) {
            // B fragment (32x16): lane=column `m`, k = half*8 + 0..7, then 16 + half*8 + 0..7
            Frag16 bf;
            const unsigned short* bp = &bBuf[buf][m * D_ + c * 32 + half * 8];
            bf.q[0] = *(const uint4*)bp;
            bf.q[1] = *(const uint4*)(bp + 16);
            acc0 = __builtin_amdgcn_wmma_f32_16x16x32_bf16(
                false, a0[c].v, false, bf.v, (short)0, acc0, false, false);
            acc1 = __builtin_amdgcn_wmma_f32_16x16x32_bf16(
                false, a1[c].v, false, bf.v, (short)0, acc1, false, false);
        }
        const float en = ldsEN[code];
        #pragma unroll
        for (int r = 0; r < 8; ++r) {
            float s0 = en - 2.0f * acc0[r];   // ||e||^2 - 2 z.e  (+||z||^2 later)
            float s1 = en - 2.0f * acc1[r];
            if (s0 < bestV0[r]) { bestV0[r] = s0; bestI0[r] = code; }
            if (s1 < bestV1[r]) { bestV1[r] = s1; bestI1[r] = code; }
        }
        buf ^= 1;
    }

    // ---- reduce argmin across the 16 lanes holding each C row ----
    const int rowBase = n0 + wv * ROWS_PER_WAVE;
    emit_rows(bestV0, bestI0, znRow0, rowBase,      lane, idxOut, counts, &sLoss);
    emit_rows(bestV1, bestI1, znRow1, rowBase + 16, lane, idxOut, counts, &sLoss);

    __syncthreads();
    if (tid == 0) atomicAdd(lossAcc, sLoss);
}

// ---------------------------------------------------------------------------
// 3) scatter: z_out[b][d][t] = emb_f32[idx[b*T+t]][d]   (coalesced writes)
// ---------------------------------------------------------------------------
__global__ void vq_scatter(const float* __restrict__ emb,
                           const int* __restrict__ idx,
                           float* __restrict__ out) {
    int i = blockIdx.x * 256 + threadIdx.x;     // 0 .. 16777215
    int t = i & (T_ - 1);
    int rest = i >> 11;
    int d = rest & (D_ - 1);
    int b = rest >> 8;
    int n = b * T_ + t;
    out[i] = emb[idx[n] * D_ + d];
}

// ---------------------------------------------------------------------------
// 4) finalize: losses + perplexity into d_out tail
// ---------------------------------------------------------------------------
__global__ void vq_finalize(const float* __restrict__ counts,
                            const float* __restrict__ lossAcc,
                            float* __restrict__ outScalars) {
    __shared__ float red[K_];
    int t = threadIdx.x;
    float p = counts[t] * (1.0f / (float)N_);
    red[t] = p * __logf(p + 1e-10f);
    __syncthreads();
    for (int off = K_ / 2; off > 0; off >>= 1) {
        if (t < off) red[t] += red[t + off];
        __syncthreads();
    }
    if (t == 0) {
        float plx  = __expf(-red[0]);
        float loss = *lossAcc * (1.0f / (float)N_);
        outScalars[0] = loss;   // z_qut_loss
        outScalars[1] = loss;   // z_enc_loss (same forward value)
        outScalars[2] = plx;    // perplexity
    }
}

// ---------------------------------------------------------------------------
// launch
// ---------------------------------------------------------------------------
extern "C" void kernel_launch(void* const* d_in, const int* in_sizes, int n_in,
                              void* d_out, int out_size, void* d_ws, size_t ws_size,
                              hipStream_t stream) {
    (void)in_sizes; (void)n_in; (void)out_size; (void)ws_size;
    const float* z   = (const float*)d_in[0];
    const float* emb = (const float*)d_in[1];
    float* out = (float*)d_out;

    // ws layout
    char* ws = (char*)d_ws;
    unsigned short* emb_bf = (unsigned short*)ws;                 // 512 KB
    float* e_norm  = (float*)(ws + 524288);                       // 4 KB
    int*   idx     = (int*)  (ws + 528384);                       // 256 KB
    float* counts  = (float*)(ws + 790528);                       // 4 KB
    float* lossAcc = (float*)(ws + 794624);                       // 4 B

    vq_init     <<<1, K_, 0, stream>>>(counts, lossAcc);
    vq_prep_emb <<<K_, D_, 0, stream>>>(emb, emb_bf, e_norm);
    vq_main     <<<N_ / TILE_ROWS, THREADS, 0, stream>>>(z, emb_bf, e_norm,
                                                         idx, counts, lossAcc);
    vq_scatter  <<<OUT_MAIN / 256, 256, 0, stream>>>(emb, idx, out);
    vq_finalize <<<1, K_, 0, stream>>>(counts, lossAcc, out + OUT_MAIN);
}
